// R2LAttentionPlusFFN_3143916061262
// MI455X (gfx1250) — compile-verified, matrix operand order checked by reading, not run on hardware
//
#include <hip/hip_runtime.h>
#include <hip/hip_bf16.h>
#include <math.h>

typedef _Float16 half_t;
typedef __attribute__((ext_vector_type(16))) _Float16 v16h;
typedef __attribute__((ext_vector_type(8)))  float    v8f;

#define NWIN 3136           // 16 * 14 * 14 windows
#define SEQ  50             // 7*7 + 1 tokens per window
#define CDIM 256
#define NH   8
#define TOK  (NWIN * SEQ)   // 156800 tokens

// ---------------------------------------------------------------------------
// WMMA fragment load from LDS.
// A 16x32 f16 fragment (and mirrored B 32x16): lanes 0-15 hold row/col = lane,
// K = 0..7 (VGPR0-3) and 16..23 (VGPR4-7); lanes 16-31 hold K+8.
// rowBase points at the K-contiguous row (or B column) start.
// ---------------------------------------------------------------------------
static __device__ __forceinline__ v16h frag_ld(const half_t* rowBase, int lane) {
  const half_t* base = rowBase + ((lane >> 4) << 3);
  v16h f;
#pragma unroll
  for (int j = 0; j < 8; ++j) {
    int k = 2 * j + ((j & 4) ? 8 : 0);
    f[2 * j]     = base[k];
    f[2 * j + 1] = base[k + 1];
  }
  return f;
}

// ---------------------------------------------------------------------------
// Elementwise helpers
// ---------------------------------------------------------------------------
__global__ void f32_to_f16_kernel(const float* __restrict__ s, half_t* __restrict__ d, int n) {
  int i = blockIdx.x * blockDim.x + threadIdx.x;
  if (i < n) d[i] = (half_t)s[i];
}

__global__ void copy_f32_kernel(const float4* __restrict__ s, float4* __restrict__ d, int n4) {
  int i = blockIdx.x * blockDim.x + threadIdx.x;
  if (i < n4) d[i] = s[i];
}

// LayerNorm over C=256, one wave per row, output f16 (compact rows of 256).
__global__ void ln_kernel(const float* __restrict__ src, long rowStride, int rows,
                          const float* __restrict__ g, const float* __restrict__ b,
                          half_t* __restrict__ dst) {
  int row = blockIdx.x * (blockDim.x >> 5) + (threadIdx.x >> 5);
  if (row >= rows) return;
  int lane = threadIdx.x & 31;
  const float4* p = (const float4*)(src + (long)row * rowStride);
  float4 a = p[lane];
  float4 c = p[lane + 32];
  float s  = a.x + a.y + a.z + a.w + c.x + c.y + c.z + c.w;
  float s2 = a.x * a.x + a.y * a.y + a.z * a.z + a.w * a.w +
             c.x * c.x + c.y * c.y + c.z * c.z + c.w * c.w;
#pragma unroll
  for (int off = 16; off > 0; off >>= 1) {
    s  += __shfl_xor(s, off, 32);
    s2 += __shfl_xor(s2, off, 32);
  }
  float mean = s * (1.0f / CDIM);
  float rstd = rsqrtf(s2 * (1.0f / CDIM) - mean * mean + 1e-5f);
  float4 ga = ((const float4*)g)[lane], gc = ((const float4*)g)[lane + 32];
  float4 ba = ((const float4*)b)[lane], bc = ((const float4*)b)[lane + 32];
  half_t* d = dst + (long)row * CDIM;
  int c0 = lane * 4, c1 = 128 + lane * 4;
  d[c0 + 0] = (half_t)((a.x - mean) * rstd * ga.x + ba.x);
  d[c0 + 1] = (half_t)((a.y - mean) * rstd * ga.y + ba.y);
  d[c0 + 2] = (half_t)((a.z - mean) * rstd * ga.z + ba.z);
  d[c0 + 3] = (half_t)((a.w - mean) * rstd * ga.w + ba.w);
  d[c1 + 0] = (half_t)((c.x - mean) * rstd * gc.x + bc.x);
  d[c1 + 1] = (half_t)((c.y - mean) * rstd * gc.y + bc.y);
  d[c1 + 2] = (half_t)((c.z - mean) * rstd * gc.z + bc.z);
  d[c1 + 3] = (half_t)((c.w - mean) * rstd * gc.w + bc.w);
}

// ---------------------------------------------------------------------------
// WMMA GEMM:  out[M][NDIM] = A[M][K](f16) * W[NDIM][K]^T(f16) (+bias)(+gelu)(+resid)
// 64x64 block tile, 128 threads (4 waves), each wave a 32x32 tile (2x2 WMMA).
// Tiles are filled with ASYNC global->LDS (TDM-path) loads, double buffered:
// while WMMA consumes buffer b, buffer b^1 is being filled (ASYNCcnt-tracked).
// ---------------------------------------------------------------------------
__global__ __launch_bounds__(128)
void gemm_kernel(const half_t* __restrict__ A, int K,
                 const half_t* __restrict__ W, int NDIM,
                 const float* __restrict__ bias,
                 const float* __restrict__ resid, long residStride,
                 float* __restrict__ outF, half_t* __restrict__ outH, long outStride,
                 int M, int act) {
  __shared__ half_t sA[2][64 * 40];
  __shared__ half_t sB[2][64 * 40];
  int tid  = threadIdx.x;
  int lane = tid & 31, wv = tid >> 5;
  int mBase = blockIdx.x * 64, nBase = blockIdx.y * 64;
  int waveM = (wv & 1) * 32, waveN = (wv >> 1) * 32;

  v8f zv = {0.f, 0.f, 0.f, 0.f, 0.f, 0.f, 0.f, 0.f};
  v8f acc[2][2];
  acc[0][0] = zv; acc[0][1] = zv; acc[1][0] = zv; acc[1][1] = zv;

  // Each thread owns one 16-half unit (32B) of the A tile and of the B tile.
  int rowU = tid >> 1, partU = tid & 1;
  // Clamp out-of-range rows instead of predicating: keeps the per-wave ASYNCcnt
  // issue count uniform; garbage lands only in discarded D rows.
  int rA = mBase + rowU; if (rA >= M)    rA = M - 1;
  int rB = nBase + rowU; if (rB >= NDIM) rB = NDIM - 1;
  const half_t* gA0 = A + (long)rA * K + partU * 16;
  const half_t* gB0 = W + (long)rB * K + partU * 16;
  unsigned idx = (unsigned)(rowU * 40 + partU * 16);
  // Flat pointers to LDS: low 32 bits are the wave-relative LDS byte address.
  unsigned lA[2] = { (unsigned)(uintptr_t)&sA[0][idx], (unsigned)(uintptr_t)&sA[1][idx] };
  unsigned lB[2] = { (unsigned)(uintptr_t)&sB[0][idx], (unsigned)(uintptr_t)&sB[1][idx] };

  auto issue_tile = [&](int buf, int k0) {
    unsigned long long pa = (unsigned long long)(uintptr_t)(gA0 + k0);
    unsigned long long pb = (unsigned long long)(uintptr_t)(gB0 + k0);
    // INST_OFFSET is added to BOTH the global and the LDS address.
    asm volatile("global_load_async_to_lds_b128 %0, %1, off"           :: "v"(lA[buf]), "v"(pa) : "memory");
    asm volatile("global_load_async_to_lds_b128 %0, %1, off offset:16" :: "v"(lA[buf]), "v"(pa) : "memory");
    asm volatile("global_load_async_to_lds_b128 %0, %1, off"           :: "v"(lB[buf]), "v"(pb) : "memory");
    asm volatile("global_load_async_to_lds_b128 %0, %1, off offset:16" :: "v"(lB[buf]), "v"(pb) : "memory");
  };

  issue_tile(0, 0);
  int buf = 0;
  for (int k0 = 0; k0 < K; k0 += 32) {
    if (k0 + 32 < K) {
      issue_tile(buf ^ 1, k0 + 32);
      // 4 in-flight from the prefetch group; wait for the consumed group only.
      asm volatile("s_wait_asynccnt 4" ::: "memory");
    } else {
      asm volatile("s_wait_asynccnt 0" ::: "memory");
    }
    __syncthreads();
    const half_t* cA = sA[buf];
    const half_t* cB = sB[buf];
    v16h af0 = frag_ld(&cA[(waveM +      (lane & 15)) * 40], lane);
    v16h af1 = frag_ld(&cA[(waveM + 16 + (lane & 15)) * 40], lane);
    v16h bf0 = frag_ld(&cB[(waveN +      (lane & 15)) * 40], lane);
    v16h bf1 = frag_ld(&cB[(waveN + 16 + (lane & 15)) * 40], lane);
    acc[0][0] = __builtin_amdgcn_wmma_f32_16x16x32_f16(false, af0, false, bf0, (short)0, acc[0][0], false, false);
    acc[0][1] = __builtin_amdgcn_wmma_f32_16x16x32_f16(false, af0, false, bf1, (short)0, acc[0][1], false, false);
    acc[1][0] = __builtin_amdgcn_wmma_f32_16x16x32_f16(false, af1, false, bf0, (short)0, acc[1][0], false, false);
    acc[1][1] = __builtin_amdgcn_wmma_f32_16x16x32_f16(false, af1, false, bf1, (short)0, acc[1][1], false, false);
    __syncthreads();
    buf ^= 1;
  }

#pragma unroll
  for (int mi = 0; mi < 2; ++mi) {
#pragma unroll
    for (int ni = 0; ni < 2; ++ni) {
      int col = nBase + waveN + ni * 16 + (lane & 15);
      float bv = bias ? bias[col] : 0.0f;
      int r0 = mBase + waveM + mi * 16 + ((lane >> 4) << 3);
#pragma unroll
      for (int i = 0; i < 8; ++i) {
        int r = r0 + i;
        if (r >= M) continue;
        float v = acc[mi][ni][i] + bv;
        if (act == 1) v = 0.5f * v * (1.0f + erff(v * 0.70710678118654752f));
        if (resid) v += resid[(long)r * residStride + col];
        if (outF) outF[(long)r * outStride + col] = v;
        if (outH) outH[(long)r * outStride + col] = (half_t)v;
      }
    }
  }
}

// ---------------------------------------------------------------------------
// WMMA attention: one block = one (item, head, 64-query-row strip).
// qkv rows: item*L .. item*L+L-1, each row [3][8][32] f16. Output o: [rows][256] f16.
// Scores via Q@K^T (K=32, one WMMA per 16x16 tile), softmax (+rel_pos/+mask) in LDS,
// then P@V via WMMA with V stored transposed in LDS.
// ---------------------------------------------------------------------------
__global__ __launch_bounds__(128)
void attn_kernel(const half_t* __restrict__ qkv, half_t* __restrict__ o,
                 const float* __restrict__ rel_pos, const int* __restrict__ mask,
                 int L, int Lpad, int qBlocks) {
  extern __shared__ char smem[];
  const int SSTR = Lpad + 1;       // score row stride (floats)
  const int VTSTR = Lpad + 8;      // transposed-V row stride (halves)
  const int PSTR = 2 * SSTR;       // f16 P row stride, aliases sS
  half_t* sQ  = (half_t*)smem;                 // 64 x 40
  half_t* sK  = sQ + 64 * 40;                  // Lpad x 40
  half_t* sVt = sK + Lpad * 40;                // 32 x VTSTR (d-major)
  float*  sS  = (float*)(sVt + 32 * VTSTR);    // 64 x SSTR
  half_t* sP  = (half_t*)sS;                   // in-place f16 probabilities

  int bid = blockIdx.x;
  int qb = bid % qBlocks; bid /= qBlocks;
  int h = bid % NH; int item = bid / NH;
  int q0 = qb * 64;
  long rowBase = (long)item * L;
  int tid = threadIdx.x, lane = tid & 31, wv = tid >> 5;

  // --- load Q (64 rows x 32 halves, zero-padded) ---
  for (int u = tid; u < 256; u += 128) {
    int r = u >> 2, part = u & 3;
    uint4 v; v.x = v.y = v.z = v.w = 0u;
    int qi = q0 + r;
    if (qi < L) v = *(const uint4*)(qkv + (rowBase + qi) * 768 + h * 32 + part * 8);
    *(uint4*)&sQ[r * 40 + part * 8] = v;
  }
  // --- load K row-major and V transposed (d-major), zero-padded ---
  for (int u = tid; u < Lpad * 4; u += 128) {
    int key = u >> 2, part = u & 3;
    uint4 kv; kv.x = kv.y = kv.z = kv.w = 0u;
    union { uint4 q; half_t e[8]; } tv; tv.q.x = tv.q.y = tv.q.z = tv.q.w = 0u;
    if (key < L) {
      kv   = *(const uint4*)(qkv + (rowBase + key) * 768 + 256 + h * 32 + part * 8);
      tv.q = *(const uint4*)(qkv + (rowBase + key) * 768 + 512 + h * 32 + part * 8);
    }
    *(uint4*)&sK[key * 40 + part * 8] = kv;
#pragma unroll
    for (int e = 0; e < 8; ++e) sVt[(part * 8 + e) * VTSTR + key] = tv.e[e];
  }
  __syncthreads();

  // --- scores: wave wv handles query rows [16*wv, 16*wv+16) of this strip ---
  const float scale = 0.17677669529663687f;  // 32^-0.5
  v8f zv = {0.f, 0.f, 0.f, 0.f, 0.f, 0.f, 0.f, 0.f};
  v16h aq = frag_ld(&sQ[(wv * 16 + (lane & 15)) * 40], lane);
  int nTiles = Lpad >> 4;
  for (int j = 0; j < nTiles; ++j) {
    v16h bk = frag_ld(&sK[(j * 16 + (lane & 15)) * 40], lane);
    v8f s = __builtin_amdgcn_wmma_f32_16x16x32_f16(false, aq, false, bk, (short)0, zv, false, false);
    int col = j * 16 + (lane & 15);
    int lm0 = wv * 16 + ((lane >> 4) << 3);
#pragma unroll
    for (int i = 0; i < 8; ++i) {
      int lm = lm0 + i;
      int qi = q0 + lm;
      float v = s[i] * scale;
      if (col >= L) {
        v = -3.0e38f;  // padded key -> softmax zero
      } else if (qi < L) {
        if (rel_pos && qi >= 1 && col >= 1) {
          int a = qi - 1, c = col - 1;
          int idx = ((a / 7 - c / 7) + 6) * 13 + ((a % 7 - c % 7) + 6);
          v += rel_pos[h * 169 + idx];
        }
        if (mask && mask[(long)item * L * L + (long)qi * L + col] == 0) v += -1.0e9f;
      }
      sS[lm * SSTR + col] = v;
    }
  }
  __syncthreads();

  // --- softmax per row; write P as f16 in place (regs staged per row) ---
  int nc = Lpad >> 5;  // values per lane (<=7)
  for (int rr = 0; rr < 16; ++rr) {
    int lm = wv * 16 + rr;
    float vals[8];
    float mx = -3.0e38f;
    for (int t = 0; t < nc; ++t) { vals[t] = sS[lm * SSTR + lane + 32 * t]; mx = fmaxf(mx, vals[t]); }
#pragma unroll
    for (int off = 16; off > 0; off >>= 1) mx = fmaxf(mx, __shfl_xor(mx, off, 32));
    float sum = 0.f;
    for (int t = 0; t < nc; ++t) { vals[t] = __expf(vals[t] - mx); sum += vals[t]; }
#pragma unroll
    for (int off = 16; off > 0; off >>= 1) sum += __shfl_xor(sum, off, 32);
    float inv = 1.0f / sum;
    for (int t = 0; t < nc; ++t) sP[lm * PSTR + lane + 32 * t] = (half_t)(vals[t] * inv);
  }
  __syncthreads();

  // --- P @ V : 16 rows per wave, 32 output dims, K = Lpad in chunks of 32 ---
  v8f oacc[2]; oacc[0] = zv; oacc[1] = zv;
  int kcN = Lpad >> 5;
  for (int kc = 0; kc < kcN; ++kc) {
    v16h ap = frag_ld(&sP[(wv * 16 + (lane & 15)) * PSTR + kc * 32], lane);
#pragma unroll
    for (int ni = 0; ni < 2; ++ni) {
      v16h bv = frag_ld(&sVt[(ni * 16 + (lane & 15)) * VTSTR + kc * 32], lane);
      oacc[ni] = __builtin_amdgcn_wmma_f32_16x16x32_f16(false, ap, false, bv, (short)0, oacc[ni], false, false);
    }
  }
#pragma unroll
  for (int ni = 0; ni < 2; ++ni) {
    int d = ni * 16 + (lane & 15);
    int lm0 = wv * 16 + ((lane >> 4) << 3);
#pragma unroll
    for (int i = 0; i < 8; ++i) {
      int qi = q0 + lm0 + i;
      if (qi < L) o[(rowBase + qi) * 256 + h * 32 + d] = (half_t)oacc[ni][i];
    }
  }
}

// ---------------------------------------------------------------------------
// Host launcher
// ---------------------------------------------------------------------------
extern "C" void kernel_launch(void* const* d_in, const int* in_sizes, int n_in,
                              void* d_out, int out_size, void* d_ws, size_t ws_size,
                              hipStream_t stream) {
  (void)in_sizes; (void)n_in; (void)out_size; (void)ws_size;
  const float* x     = (const float*)d_in[0];
  const int*   mask  = (const int*)d_in[4];
  const float* g0 = (const float*)d_in[5],  *b0 = (const float*)d_in[6];
  const float* g1 = (const float*)d_in[7],  *b1 = (const float*)d_in[8];
  const float* g2 = (const float*)d_in[9],  *b2 = (const float*)d_in[10];
  const float* Wqkv  = (const float*)d_in[11];
  const float* Wproj = (const float*)d_in[12];
  const float* bproj = (const float*)d_in[13];
  const float* relp  = (const float*)d_in[14];
  const float* W1  = (const float*)d_in[15], *b1m = (const float*)d_in[16];
  const float* W2  = (const float*)d_in[17], *b2m = (const float*)d_in[18];
  float* Y = (float*)d_out;

  char* ws = (char*)d_ws;
  auto alloc = [&](size_t bytes) -> char* {
    char* p = ws; ws += (bytes + 255) & ~(size_t)255; return p;
  };
  half_t* wqkv16  = (half_t*)alloc((size_t)768 * 256 * 2);
  half_t* wproj16 = (half_t*)alloc((size_t)256 * 256 * 2);
  half_t* w116    = (half_t*)alloc((size_t)1024 * 256 * 2);
  half_t* w216    = (half_t*)alloc((size_t)256 * 1024 * 2);
  half_t* clsln   = (half_t*)alloc((size_t)NWIN * 256 * 2);
  half_t* qkvcls  = (half_t*)alloc((size_t)NWIN * 768 * 2);
  half_t* xln     = (half_t*)alloc((size_t)TOK * 256 * 2);
  half_t* obuf    = (half_t*)alloc((size_t)TOK * 256 * 2);
  half_t* qkvb    = (half_t*)alloc((size_t)TOK * 768 * 2);
  const long CH = 616 * 64;  // token chunk for FFN scratch
  half_t* hbuf    = (half_t*)alloc((size_t)CH * 1024 * 2);

  // weights -> f16
  f32_to_f16_kernel<<<(768 * 256 + 255) / 256, 256, 0, stream>>>(Wqkv, wqkv16, 768 * 256);
  f32_to_f16_kernel<<<(256 * 256 + 255) / 256, 256, 0, stream>>>(Wproj, wproj16, 256 * 256);
  f32_to_f16_kernel<<<(1024 * 256 + 255) / 256, 256, 0, stream>>>(W1, w116, 1024 * 256);
  f32_to_f16_kernel<<<(256 * 1024 + 255) / 256, 256, 0, stream>>>(W2, w216, 256 * 1024);

  // Y = x (running residual)
  int n4 = TOK * 256 / 4;
  copy_f32_kernel<<<(n4 + 255) / 256, 256, 0, stream>>>((const float4*)x, (float4*)Y, n4);

  // ---- cls attention over (16 batches, 196 cls tokens) ----
  ln_kernel<<<(NWIN + 7) / 8, 256, 0, stream>>>(x, (long)SEQ * 256, NWIN, g0, b0, clsln);
  gemm_kernel<<<dim3(NWIN / 64, 768 / 64), 128, 0, stream>>>(
      clsln, 256, wqkv16, 768, nullptr, nullptr, 0, nullptr, qkvcls, 768, NWIN, 0);
  {
    int L = 196, Lpad = 224, qB = 4;
    size_t sh = (size_t)(64 * 40 + Lpad * 40 + 32 * (Lpad + 8)) * 2 + (size_t)64 * (Lpad + 1) * 4;
    (void)hipFuncSetAttribute((const void*)attn_kernel,
                              hipFuncAttributeMaxDynamicSharedMemorySize, (int)sh);
    attn_kernel<<<16 * NH * qB, 128, sh, stream>>>(qkvcls, obuf, nullptr, nullptr, L, Lpad, qB);
  }
  // proj + residual back into Y[:, 0, :] (row stride SEQ*256)
  gemm_kernel<<<dim3(NWIN / 64, 256 / 64), 128, 0, stream>>>(
      obuf, 256, wproj16, 256, bproj, Y, (long)SEQ * 256, Y, nullptr, (long)SEQ * 256, NWIN, 0);

  // ---- window attention over all tokens ----
  ln_kernel<<<(TOK + 7) / 8, 256, 0, stream>>>(Y, 256, TOK, g1, b1, xln);
  gemm_kernel<<<dim3(TOK / 64, 768 / 64), 128, 0, stream>>>(
      xln, 256, wqkv16, 768, nullptr, nullptr, 0, nullptr, qkvb, 768, TOK, 0);
  {
    int L = SEQ, Lpad = 64, qB = 1;
    size_t sh = (size_t)(64 * 40 + Lpad * 40 + 32 * (Lpad + 8)) * 2 + (size_t)64 * (Lpad + 1) * 4;
    attn_kernel<<<NWIN * NH, 128, sh, stream>>>(qkvb, obuf, relp, mask, L, Lpad, qB);
  }
  gemm_kernel<<<dim3(TOK / 64, 256 / 64), 128, 0, stream>>>(
      obuf, 256, wproj16, 256, bproj, Y, 256, Y, nullptr, 256, TOK, 0);

  // ---- FFN: gelu(LN(Y) @ W1^T + b1) @ W2^T + b2 + Y, chunked over tokens ----
  ln_kernel<<<(TOK + 7) / 8, 256, 0, stream>>>(Y, 256, TOK, g2, b2, xln);
  for (long ofs = 0; ofs < TOK; ofs += CH) {
    int Mc = (int)((TOK - ofs < CH) ? (TOK - ofs) : CH);
    int gm = (Mc + 63) / 64;
    gemm_kernel<<<dim3(gm, 1024 / 64), 128, 0, stream>>>(
        xln + ofs * 256, 256, w116, 1024, b1m, nullptr, 0, nullptr, hbuf, 1024, Mc, 1);
    gemm_kernel<<<dim3(gm, 256 / 64), 128, 0, stream>>>(
        hbuf, 1024, w216, 256, b2m, Y + ofs * 256, 256, Y + ofs * 256, nullptr, 256, Mc, 0);
  }
}
